// InpWithDistEncoder_69801808495247
// MI455X (gfx1250) — matrix operand order repaired and checked by reading, not compiled, not run on hardware
//
#include <hip/hip_runtime.h>
#include <hip/hip_bf16.h>
#include <math.h>

// ---------------------------------------------------------------------------
// InpWithDistEncoder for MI455X (gfx1250, wave32, WMMA)
//   out[b,l,:] = concat( dist(1) | word_emb(300) | char_cnn(100) | pos_emb(100) )
// ---------------------------------------------------------------------------

#define BS 64
#define LS 512
#define CL 32          // chars per token
#define CD 100         // char embedding dim (GEMM K, padded to 128)
#define NF 100         // conv filters
#define KK 3           // conv kernel taps
#define OUTD 501
#define NTOK (BS * LS) // 32768 tokens

#define KPAD 128       // K padded: 4 chunks of 32
#define NFPAD 112      // filters padded per tap: 7 tiles of 16
#define NB (KK * NFPAD) // 336 B-matrix columns

typedef __attribute__((ext_vector_type(16))) _Float16 v16h;
typedef __attribute__((ext_vector_type(8)))  float    v8f;
typedef __attribute__((ext_vector_type(4)))  float    v4f;
typedef __attribute__((ext_vector_type(4)))  unsigned int uint4v;

union Frag16 { v16h h; uint4v u[2]; };
union Row32  { v4f v[8]; float f[32]; };

// LDS layout
#define CT_HALVES  (100 * KPAD)         // 12800 f16: char table, row=char id, K-major
#define WB_HALVES  (NB * KPAD)          // 43008 f16: weights, row n = kk*112+f, col = channel
#define ZST_WAVE_FLOATS (KK * 16 * 32)  // 1536 f32 per wave: [kk][col16][row32]
#define SMEM_BYTES ((CT_HALVES + WB_HALVES) * 2 + 8 * ZST_WAVE_FLOATS * 4) // 160768 B

// ---------------------------------------------------------------------------
// Kernel 1: nearest-privacy-token distance feature -> out[...,0]
// ---------------------------------------------------------------------------
__global__ void dist_kernel(const int* __restrict__ input_pos,
                            float* __restrict__ out) {
  __shared__ int mask[LS];
  const int b = blockIdx.x;
  const int j = threadIdx.x;
  const int p = input_pos[b * LS + j];
  mask[j] = (p == 3) | (p == 7) | (p == 11);
  __syncthreads();
  int last = -1000, nxt = 1000;
  for (int t = 0; t < LS; ++t) {
    if (mask[t]) {
      if (t <= j) last = t;
      if (t >= j && nxt == 1000) nxt = t;
    }
  }
  const int d = min(j - last, nxt - j);
  out[(size_t)(b * LS + j) * OUTD] = (float)d;
}

// ---------------------------------------------------------------------------
// Kernel 2: word + pos embedding gather (memory-bound, wave-per-token)
// ---------------------------------------------------------------------------
__global__ void gather_kernel(const int* __restrict__ input_word,
                              const int* __restrict__ input_pos,
                              const float* __restrict__ word_table,
                              const float* __restrict__ pos_table,
                              float* __restrict__ out) {
  const int wv = threadIdx.x >> 5;
  const int lane = threadIdx.x & 31;
  const int t = blockIdx.x * 8 + wv;
  if (t >= NTOK) return;
  const int widx = input_word[t];
  const int pidx = input_pos[t];
  float* o = out + (size_t)t * OUTD;
  const float* wr = word_table + (size_t)widx * 300;
  for (int c = lane; c < 300; c += 32) o[1 + c] = wr[c];
  const float* pr = pos_table + (size_t)pidx * 100;
  for (int c = lane; c < 100; c += 32) o[401 + c] = pr[c];
}

// ---------------------------------------------------------------------------
// Kernel 3: char CNN via v_wmma_f32_16x16x32_f16
//   Per token (one wave): A = char embeddings [32 x 100] (K padded to 128),
//   B = weights reshaped [100 x 336] with col n = kk*112 + f,
//   Z_kk[i,f] = sum_c A[i,c]*w[f,c,kk];  Y[o,f] = sum_kk Z_kk[o+kk-2,f];
//   out = tanh(max_o Y + bias).
//   Z staged f32, column-major [kk][col16][row32] -> b128 stores/loads.
// ---------------------------------------------------------------------------
__global__ void __launch_bounds__(256)
charcnn_kernel(const int* __restrict__ input_char,
               const float* __restrict__ char_table,
               const float* __restrict__ conv_w,
               const float* __restrict__ conv_b,
               float* __restrict__ out) {
  extern __shared__ __align__(16) _Float16 smem[];
  _Float16* ct  = smem;                 // [100][128] f16
  _Float16* wb  = smem + CT_HALVES;     // [336][128] f16
  float*    zst = (float*)(smem + CT_HALVES + WB_HALVES); // [8][3][16][32] f32

  // --- Stage tables into LDS (f16, zero-padded) ---
  for (int idx = threadIdx.x; idx < CT_HALVES; idx += 256) {
    const int c = idx >> 7, d = idx & 127;
    ct[idx] = (d < CD) ? (_Float16)char_table[c * CD + d] : (_Float16)0.f;
  }
  for (int idx = threadIdx.x; idx < WB_HALVES; idx += 256) {
    const int n = idx >> 7, cc = idx & 127;
    const int kk = n / NFPAD, f = n % NFPAD;
    float v = 0.f;
    if (f < NF && cc < CD) v = conv_w[(f * CD + cc) * KK + kk];
    wb[idx] = (_Float16)v;
  }
  __syncthreads();

  const int wv   = threadIdx.x >> 5;
  const int lane = threadIdx.x & 31;
  const int aOff = (lane < 16) ? 0 : 8;   // A fragment half-offset
  const int bK   = (lane < 16) ? 0 : 8;   // B fragment K-offset
  const int nCol = lane & 15;
  const int r0   = (lane < 16) ? 0 : 8;   // D-tile row base per lane half
  float* zw = zst + wv * ZST_WAVE_FLOATS;

  // 512 blocks * 8 waves * 8 rounds == 32768 tokens exactly (no partial waves)
  for (int t = blockIdx.x * 8 + wv; t < NTOK; t += gridDim.x * 8) {
    const int* cptr = input_char + (size_t)t * CL;

    // --- Load A fragments: char embeddings, all K chunks, both M tiles ---
    Frag16 a[2][4];
#pragma unroll
    for (int mt = 0; mt < 2; ++mt) {
      const int i = mt * 16 + (lane & 15);
      const int cidx = cptr[i];
      const _Float16* crow = ct + cidx * KPAD;
#pragma unroll
      for (int kc = 0; kc < 4; ++kc) {
        a[mt][kc].u[0] = *(const uint4v*)(crow + kc * 32 + aOff);
        a[mt][kc].u[1] = *(const uint4v*)(crow + kc * 32 + 16 + aOff);
      }
    }

    // --- 7 filter tiles; for each, compute 3 tap-tiles x 2 M tiles ---
    for (int ft = 0; ft < 7; ++ft) {
#pragma unroll
      for (int kk = 0; kk < KK; ++kk) {
        const _Float16* wrow = wb + (size_t)(kk * NFPAD + ft * 16 + nCol) * KPAD;
        v8f acc0 = {0.f, 0.f, 0.f, 0.f, 0.f, 0.f, 0.f, 0.f};
        v8f acc1 = {0.f, 0.f, 0.f, 0.f, 0.f, 0.f, 0.f, 0.f};
#pragma unroll
        for (int kc = 0; kc < 4; ++kc) {
          Frag16 bf;
          bf.u[0] = *(const uint4v*)(wrow + kc * 32 + bK);
          bf.u[1] = *(const uint4v*)(wrow + kc * 32 + 16 + bK);
          acc0 = __builtin_amdgcn_wmma_f32_16x16x32_f16(
              false, a[0][kc].h, false, bf.h, (short)0, acc0, false, false);
          acc1 = __builtin_amdgcn_wmma_f32_16x16x32_f16(
              false, a[1][kc].h, false, bf.h, (short)0, acc1, false, false);
        }
        // --- Stage Z tile to per-wave LDS, f32 column-major (b128 stores) ---
        float* zc = zw + (kk * 16 + nCol) * 32;
        v4f s0 = {acc0[0], acc0[1], acc0[2], acc0[3]};
        v4f s1 = {acc0[4], acc0[5], acc0[6], acc0[7]};
        v4f s2 = {acc1[0], acc1[1], acc1[2], acc1[3]};
        v4f s3 = {acc1[4], acc1[5], acc1[6], acc1[7]};
        *(v4f*)(zc + r0)          = s0;   // mt0 rows r0..r0+3
        *(v4f*)(zc + r0 + 4)      = s1;   // mt0 rows r0+4..r0+7
        *(v4f*)(zc + 16 + r0)     = s2;   // mt1 rows 16+r0..
        *(v4f*)(zc + 16 + r0 + 4) = s3;
      }

      asm volatile("s_wait_dscnt 0" ::: "memory");

      // --- Combine: pull 3x32-row windows (b128), shift-add taps, max, tanh ---
      const int fl = lane & 15;
      const float* z0 = zw + (0 * 16 + fl) * 32;
      const float* z1 = zw + (1 * 16 + fl) * 32;
      const float* z2 = zw + (2 * 16 + fl) * 32;
      Row32 R0, R1, R2;
#pragma unroll
      for (int q = 0; q < 8; ++q) {
        R0.v[q] = *(const v4f*)(z0 + q * 4);
        R1.v[q] = *(const v4f*)(z1 + q * 4);
        R2.v[q] = *(const v4f*)(z2 + q * 4);
      }
      float m = -3.4e38f;
#pragma unroll
      for (int o = 0; o < 34; ++o) {
        float s = 0.f;
        if (o >= 2)            s += R0.f[o - 2];
        if (o >= 1 && o < 33)  s += R1.f[o - 1];
        if (o < 32)            s += R2.f[o];
        m = fmaxf(m, s);
      }
      const int f = ft * 16 + fl;
      if (lane < 16 && f < NF) {
        out[(size_t)t * OUTD + 301 + f] = tanhf(m + conv_b[f]);
      }
    }
  }
}

// ---------------------------------------------------------------------------
// Host launcher
// ---------------------------------------------------------------------------
extern "C" void kernel_launch(void* const* d_in, const int* in_sizes, int n_in,
                              void* d_out, int out_size, void* d_ws, size_t ws_size,
                              hipStream_t stream) {
  (void)in_sizes; (void)n_in; (void)out_size; (void)d_ws; (void)ws_size;
  const int*   input_word = (const int*)d_in[0];
  const int*   input_char = (const int*)d_in[1];
  const int*   input_pos  = (const int*)d_in[2];
  const float* word_table = (const float*)d_in[3];
  const float* char_table = (const float*)d_in[4];
  const float* pos_table  = (const float*)d_in[5];
  const float* conv_w     = (const float*)d_in[6];
  const float* conv_b     = (const float*)d_in[7];
  float* out = (float*)d_out;

  dist_kernel<<<BS, LS, 0, stream>>>(input_pos, out);
  gather_kernel<<<NTOK / 8, 256, 0, stream>>>(input_word, input_pos,
                                              word_table, pos_table, out);
  charcnn_kernel<<<512, 256, SMEM_BYTES, stream>>>(input_char, char_table,
                                                   conv_w, conv_b, out);
}
